// GCN_85126251807570
// MI455X (gfx1250) — matrix operand order repaired and checked by reading, not compiled, not run on hardware
//
#include <hip/hip_runtime.h>
#include <hip/hip_bf16.h>
#include <math.h>

// ---------------- problem constants ----------------
#define NN   100000
#define EE   1000000
#define CIN  128
#define HID  128
#define OUTC 40
#define OUTP 48          // padded to 3 tiles of 16
#define BN_EPS 1e-5f

// ---------------- WMMA vector types ----------------
typedef __attribute__((ext_vector_type(16))) __bf16         v16bf;
typedef __attribute__((ext_vector_type(8)))  float          v8f;

// =====================================================================
// GEMM: C[N,COUT] = A[N,128] * W[128,COUT]   (bf16 WMMA, f32 accumulate)
// One 256-thread block = 8 waves; wave handles 16 rows x COUT cols.
// W staged in LDS pre-swizzled into B-fragment lane layout.
// =====================================================================
template<int COUT>
__global__ __launch_bounds__(256) void gemm_bf16_wmma(
    const float* __restrict__ A, const float* __restrict__ W,
    float* __restrict__ C, int nrows)
{
    constexpr int NT = COUT / 16;   // n tiles
    constexpr int KT = CIN  / 32;   // 4 k tiles
    __shared__ __align__(32) __bf16 ldsB[KT * NT * 32 * 16];

    const int tid = threadIdx.x;

    // ---- stage W into LDS, swizzled per-fragment-per-lane ----
    // entry (kt, nt, L): 16 bf16 values = column n = nt*16 + L%16,
    //                    k = kt*32 + (L>=16 ? 8 : 0) + {0..7, 16..23}
    const int entries = KT * NT * 32;
    for (int ent = tid; ent < entries; ent += 256) {
        int kt  = ent / (NT * 32);
        int rem = ent % (NT * 32);
        int nt  = rem / 32;
        int L   = rem % 32;
        int kb  = kt * 32 + ((L >= 16) ? 8 : 0);
        int n   = nt * 16 + (L & 15);
        __bf16* dst = &ldsB[ent * 16];
#pragma unroll
        for (int j = 0; j < 16; ++j)
            dst[j] = (__bf16)W[(kb + (j < 8 ? j : j + 8)) * COUT + n];   // hw cvt (v_cvt_pk_bf16_f32)
    }
    __syncthreads();

    const int wave = tid >> 5;
    const int lane = tid & 31;
    const int row0 = blockIdx.x * 128 + wave * 16;
    if (row0 >= nrows) return;                 // N % 16 == 0, full tiles only
    const int m = row0 + (lane & 15);
    const int g = lane >> 4;                   // K sub-group select

    v8f acc[NT];
#pragma unroll
    for (int t = 0; t < NT; ++t) acc[t] = (v8f){0,0,0,0,0,0,0,0};

#pragma unroll
    for (int kt = 0; kt < KT; ++kt) {
        // A fragment (ISA 16-bit A 16x32 layout): elems 0..7 = K=g*8+0..7,
        // elems 8..15 = K=16+g*8+0..7 (relative to kt*32)
        const float* ap = A + (size_t)m * CIN + kt * 32;
        const float4 f0 = *(const float4*)(ap + g * 8);
        const float4 f1 = *(const float4*)(ap + g * 8 + 4);
        const float4 f2 = *(const float4*)(ap + 16 + g * 8);
        const float4 f3 = *(const float4*)(ap + 16 + g * 8 + 4);
        v16bf afrag;
        afrag[0]  = (__bf16)f0.x; afrag[1]  = (__bf16)f0.y;
        afrag[2]  = (__bf16)f0.z; afrag[3]  = (__bf16)f0.w;
        afrag[4]  = (__bf16)f1.x; afrag[5]  = (__bf16)f1.y;
        afrag[6]  = (__bf16)f1.z; afrag[7]  = (__bf16)f1.w;
        afrag[8]  = (__bf16)f2.x; afrag[9]  = (__bf16)f2.y;
        afrag[10] = (__bf16)f2.z; afrag[11] = (__bf16)f2.w;
        afrag[12] = (__bf16)f3.x; afrag[13] = (__bf16)f3.y;
        afrag[14] = (__bf16)f3.z; afrag[15] = (__bf16)f3.w;

#pragma unroll
        for (int nt = 0; nt < NT; ++nt) {
            const v16bf bfrag =
                *(const v16bf*)&ldsB[((kt * NT + nt) * 32 + lane) * 16];
            acc[nt] = __builtin_amdgcn_wmma_f32_16x16x32_bf16(
                false, afrag, false, bfrag, (short)0, acc[nt], false, false);
        }
    }

    // D layout: lanes 0-15 VGPR r -> M=r ; lanes 16-31 VGPR r -> M=r+8
#pragma unroll
    for (int nt = 0; nt < NT; ++nt) {
        const int n = nt * 16 + (lane & 15);
        const int mb = row0 + g * 8;
#pragma unroll
        for (int r = 0; r < 8; ++r)
            C[(size_t)(mb + r) * COUT + n] = acc[nt][r];
    }
}

// =====================================================================
// Graph normalization helpers
// =====================================================================
__global__ void degree_kernel(const int* __restrict__ ei, float* __restrict__ deg) {
    int e = blockIdx.x * blockDim.x + threadIdx.x;
    if (e < EE) atomicAdd(&deg[ei[EE + e]], 1.0f);
}

__global__ void dinv_kernel(float* __restrict__ deg) {
    int i = blockIdx.x * blockDim.x + threadIdx.x;
    if (i < NN) deg[i] = rsqrtf(deg[i] + 1.0f);
}

// msg scatter: thread = (edge, 4-channel chunk); lanes cover contiguous chans
template<int COUT>
__global__ void scatter_edges(const int* __restrict__ ei,
                              const float* __restrict__ dinv,
                              const float* __restrict__ hw,
                              float* __restrict__ agg)
{
    constexpr int CH = COUT / 4;
    long long t = (long long)blockIdx.x * blockDim.x + threadIdx.x;
    if (t >= (long long)EE * CH) return;
    int e  = (int)(t / CH);
    int c4 = (int)(t % CH);
    int s = ei[e];
    int d = ei[EE + e];
    float nrm = dinv[s] * dinv[d];
    const float4 v = *(const float4*)&hw[(size_t)s * COUT + c4 * 4];
    float* out = &agg[(size_t)d * COUT + c4 * 4];
    atomicAdd(out + 0, v.x * nrm);
    atomicAdd(out + 1, v.y * nrm);
    atomicAdd(out + 2, v.z * nrm);
    atomicAdd(out + 3, v.w * nrm);
}

// agg += hw * dinv^2  (self loop) + bias
template<int COUT>
__global__ void self_bias(const float* __restrict__ hw,
                          const float* __restrict__ dinv,
                          const float* __restrict__ b, int nb,
                          float* __restrict__ agg)
{
    long long idx = (long long)blockIdx.x * blockDim.x + threadIdx.x;
    if (idx >= (long long)NN * COUT) return;
    int i = (int)(idx / COUT);
    int c = (int)(idx % COUT);
    float di = dinv[i];
    float bb = (c < nb) ? b[c] : 0.0f;
    agg[idx] += hw[idx] * di * di + bb;
}

// =====================================================================
// BatchNorm (training stats) + ReLU + dropout
// =====================================================================
__global__ void bn_stats(const float* __restrict__ h,
                         float* __restrict__ sum, float* __restrict__ sumsq)
{
    int c    = threadIdx.x & 127;
    int half = threadIdx.x >> 7;
    float s = 0.f, sq = 0.f;
    for (int i = blockIdx.x * 2 + half; i < NN; i += gridDim.x * 2) {
        float v = h[(size_t)i * HID + c];
        s += v; sq += v * v;
    }
    atomicAdd(&sum[c], s);
    atomicAdd(&sumsq[c], sq);
}

__global__ void bn_finalize(const float* __restrict__ sum,
                            const float* __restrict__ sumsq,
                            const float* __restrict__ gamma,
                            const float* __restrict__ beta,
                            float* __restrict__ scale, float* __restrict__ shift)
{
    int c = threadIdx.x;
    float mu   = sum[c] * (1.0f / NN);
    float var  = sumsq[c] * (1.0f / NN) - mu * mu;
    float rstd = rsqrtf(var + BN_EPS);
    float sc   = gamma[c] * rstd;
    scale[c] = sc;
    shift[c] = beta[c] - mu * sc;
}

__global__ void norm_relu_drop(float* __restrict__ h,
                               const float* __restrict__ scale,
                               const float* __restrict__ shift,
                               unsigned int seed)
{
    long long idx = (long long)blockIdx.x * blockDim.x + threadIdx.x;
    if (idx >= (long long)NN * HID) return;
    int c = (int)(idx & 127);
    float v = h[idx] * scale[c] + shift[c];
    v = fmaxf(v, 0.0f);
    unsigned int x = (unsigned int)idx ^ seed;          // deterministic hash RNG
    x *= 0x9E3779B1u; x ^= x >> 16; x *= 0x85EBCA6Bu; x ^= x >> 13;
    h[idx] = (x & 1u) ? v * 2.0f : 0.0f;                // p = 0.5, scale 1/(1-p)
}

// =====================================================================
// Layer-3 helpers
// =====================================================================
__global__ void pad_w3(const float* __restrict__ W3, float* __restrict__ Wp) {
    int idx = blockIdx.x * blockDim.x + threadIdx.x;
    if (idx >= HID * OUTP) return;
    int k = idx / OUTP, c = idx % OUTP;
    Wp[idx] = (c < OUTC) ? W3[k * OUTC + c] : 0.0f;
}

__global__ void log_softmax_kernel(const float* __restrict__ agg,
                                   float* __restrict__ out)
{
    int i = blockIdx.x * blockDim.x + threadIdx.x;
    if (i >= NN) return;
    const float* row = &agg[(size_t)i * OUTP];
    float m = -INFINITY;
#pragma unroll
    for (int c = 0; c < OUTC; ++c) m = fmaxf(m, row[c]);
    float s = 0.f;
#pragma unroll
    for (int c = 0; c < OUTC; ++c) s += __expf(row[c] - m);
    float ls = __logf(s);
#pragma unroll
    for (int c = 0; c < OUTC; ++c) out[(size_t)i * OUTC + c] = row[c] - m - ls;
}

// =====================================================================
// Orchestration
// =====================================================================
extern "C" void kernel_launch(void* const* d_in, const int* in_sizes, int n_in,
                              void* d_out, int out_size, void* d_ws, size_t ws_size,
                              hipStream_t stream)
{
    const float* x   = (const float*)d_in[0];
    const int*   ei  = (const int*)  d_in[1];
    const float* W1  = (const float*)d_in[2];
    const float* b1  = (const float*)d_in[3];
    const float* g1  = (const float*)d_in[4];
    const float* be1 = (const float*)d_in[5];
    const float* W2  = (const float*)d_in[6];
    const float* b2  = (const float*)d_in[7];
    const float* g2  = (const float*)d_in[8];
    const float* be2 = (const float*)d_in[9];
    const float* W3  = (const float*)d_in[10];
    const float* b3  = (const float*)d_in[11];
    float* out = (float*)d_out;

    // workspace layout (floats)
    float* ws    = (float*)d_ws;
    float* dinv  = ws;                         // N
    float* hw    = dinv + NN;                  // N*128
    float* agg   = hw + (size_t)NN * HID;      // N*128  (also reused as h)
    float* sum   = agg + (size_t)NN * HID;     // 128
    float* sumsq = sum + HID;                  // 128
    float* scale = sumsq + HID;                // 128
    float* shift = scale + HID;                // 128
    float* Wpad  = shift + HID;                // 128*48

    const int T = 256;
    const int gemmBlocks = (NN + 127) / 128;
    const long long eTh128 = (long long)EE * (HID / 4);
    const long long eTh48  = (long long)EE * (OUTP / 4);
    const long long nTh128 = (long long)NN * HID;
    const long long nTh48  = (long long)NN * OUTP;

    // ---- degree / D^{-1/2} ----
    hipMemsetAsync(dinv, 0, (size_t)NN * 4, stream);
    degree_kernel<<<(EE + T - 1) / T, T, 0, stream>>>(ei, dinv);
    dinv_kernel<<<(NN + T - 1) / T, T, 0, stream>>>(dinv);

    // ================= layer 1 =================
    gemm_bf16_wmma<HID><<<gemmBlocks, T, 0, stream>>>(x, W1, hw, NN);
    hipMemsetAsync(agg, 0, (size_t)NN * HID * 4, stream);
    scatter_edges<HID><<<(int)((eTh128 + T - 1) / T), T, 0, stream>>>(ei, dinv, hw, agg);
    self_bias<HID><<<(int)((nTh128 + T - 1) / T), T, 0, stream>>>(hw, dinv, b1, HID, agg);
    hipMemsetAsync(sum, 0, 2 * HID * 4, stream);
    bn_stats<<<256, T, 0, stream>>>(agg, sum, sumsq);
    bn_finalize<<<1, HID, 0, stream>>>(sum, sumsq, g1, be1, scale, shift);
    norm_relu_drop<<<(int)((nTh128 + T - 1) / T), T, 0, stream>>>(agg, scale, shift, 0xA5F15273u);

    // ================= layer 2 =================
    gemm_bf16_wmma<HID><<<gemmBlocks, T, 0, stream>>>(agg, W2, hw, NN);
    hipMemsetAsync(agg, 0, (size_t)NN * HID * 4, stream);
    scatter_edges<HID><<<(int)((eTh128 + T - 1) / T), T, 0, stream>>>(ei, dinv, hw, agg);
    self_bias<HID><<<(int)((nTh128 + T - 1) / T), T, 0, stream>>>(hw, dinv, b2, HID, agg);
    hipMemsetAsync(sum, 0, 2 * HID * 4, stream);
    bn_stats<<<256, T, 0, stream>>>(agg, sum, sumsq);
    bn_finalize<<<1, HID, 0, stream>>>(sum, sumsq, g2, be2, scale, shift);
    norm_relu_drop<<<(int)((nTh128 + T - 1) / T), T, 0, stream>>>(agg, scale, shift, 0x3C6EF372u);

    // ================= layer 3 (padded 40 -> 48) =================
    pad_w3<<<(HID * OUTP + T - 1) / T, T, 0, stream>>>(W3, Wpad);
    gemm_bf16_wmma<OUTP><<<gemmBlocks, T, 0, stream>>>(agg, Wpad, hw, NN);
    hipMemsetAsync(agg, 0, (size_t)NN * OUTP * 4, stream);
    scatter_edges<OUTP><<<(int)((eTh48 + T - 1) / T), T, 0, stream>>>(ei, dinv, hw, agg);
    self_bias<OUTP><<<(int)((nTh48 + T - 1) / T), T, 0, stream>>>(hw, dinv, b3, OUTC, agg);
    log_softmax_kernel<<<(NN + T - 1) / T, T, 0, stream>>>(agg, out);
}
